// MultiHeadAttention_59992103190912
// MI455X (gfx1250) — matrix operand order repaired and checked by reading, not compiled, not run on hardware
//
#include <hip/hip_runtime.h>
#include <hip/hip_bf16.h>

typedef _Float16 v8h  __attribute__((ext_vector_type(8)));
typedef _Float16 v16h __attribute__((ext_vector_type(16)));
typedef float    v8f  __attribute__((ext_vector_type(8)));

#define BATCH 4
#define TSEQ  2048
#define DMODEL 1024
#define NHEAD 16
#define DHEAD 64
#define MROWS (BATCH * TSEQ)     // 8192
#define SCALE 0.125f             // 64^-0.5
#define LN_EPS 1e-5f

// ---- CDNA5 async global->LDS path (guarded; falls back to manual staging) ----
#if defined(__has_builtin)
#if __has_builtin(__builtin_amdgcn_global_load_async_to_lds_b128) && \
    __has_builtin(__builtin_amdgcn_s_wait_asynccnt)
#define HAVE_ASYNC_LDS 1
#endif
#endif
#ifndef HAVE_ASYNC_LDS
#define HAVE_ASYNC_LDS 0
#endif

#if HAVE_ASYNC_LDS
// builtin signature (from hipcc diagnostic): param0 = AS1 int4*, param1 = AS3 int4*,
// then imm offset, imm cpol.
typedef int v4i_vs __attribute__((vector_size(16)));
typedef __attribute__((address_space(1))) v4i_vs gv4i;
typedef __attribute__((address_space(3))) v4i_vs lv4i;
#define ASYNC_CP16(gptr, lptr) \
    __builtin_amdgcn_global_load_async_to_lds_b128((gv4i*)(gptr), (lv4i*)(lptr), 0, 0)
#define ASYNC_WAIT() __builtin_amdgcn_s_wait_asynccnt(0)
#else
#define ASYNC_CP16(gptr, lptr) \
    do { *(v8h*)(lptr) = *(const v8h*)(gptr); } while (0)
#define ASYNC_WAIT() do {} while (0)
#endif

// ---------------- workspace layout (bytes) ----------------
// weights stored TRANSPOSED f16: Wt[n][k]
#define OFF_WQ  (0ull)
#define OFF_WK  (OFF_WQ + 2ull*1024*1024)
#define OFF_WV  (OFF_WK + 2ull*1024*1024)
#define OFF_WO  (OFF_WV + 2ull*1024*1024)
#define OFF_XN  (OFF_WO + 2ull*1024*1024)
#define OFF_Q   (OFF_XN + 16ull*1024*1024)
#define OFF_K   (OFF_Q  + 16ull*1024*1024)
#define OFF_V   (OFF_K  + 16ull*1024*1024)   // V stored [B,H,DH,T]
#define OFF_AO  (OFF_V  + 16ull*1024*1024)

// ---------------- transpose + cast f32 -> f16: dst[n][k] = src[k][n] ----------------
// 64x64 LDS tiles; both global sides coalesced. grid = (16,16), block = 256.
__global__ __launch_bounds__(256) void transpose_cast_kernel(const float* __restrict__ src,
                                                             _Float16* __restrict__ dst) {
    __shared__ _Float16 tile[64][65];
    int k0 = blockIdx.x * 64, n0 = blockIdx.y * 64;
    int tid = threadIdx.x;
    for (int jj = 0; jj < 16; jj++) {
        int idx = tid + jj * 256;
        int r = idx >> 6, c = idx & 63;                      // r: k-offset, c: n-offset
        tile[c][r] = (_Float16)src[(size_t)(k0 + r) * DMODEL + n0 + c];
    }
    __syncthreads();
    for (int jj = 0; jj < 16; jj++) {
        int idx = tid + jj * 256;
        int rn = idx >> 6, ck = idx & 63;
        dst[(size_t)(n0 + rn) * DMODEL + k0 + ck] = tile[rn][ck];
    }
}

// ---------------- LayerNorm (one row of D=1024 per block) ----------------
__global__ __launch_bounds__(256) void ln_kernel(const float* __restrict__ x,
                                                 const float* __restrict__ gamma,
                                                 const float* __restrict__ beta,
                                                 _Float16* __restrict__ xn) {
    int row = blockIdx.x;
    int tid = threadIdx.x;
    const float4 v  = ((const float4*)(x + (size_t)row * DMODEL))[tid];
    float s  = v.x + v.y + v.z + v.w;
    float ss = v.x * v.x + v.y * v.y + v.z * v.z + v.w * v.w;
    for (int off = 16; off > 0; off >>= 1) {
        s  += __shfl_xor(s,  off, 32);
        ss += __shfl_xor(ss, off, 32);
    }
    __shared__ float rs[8], rss[8];
    int wave = tid >> 5, lane = tid & 31;
    if (lane == 0) { rs[wave] = s; rss[wave] = ss; }
    __syncthreads();
    if (tid == 0) {
        float a = 0.f, b2 = 0.f;
        for (int i = 0; i < 8; i++) { a += rs[i]; b2 += rss[i]; }
        rs[0] = a; rss[0] = b2;
    }
    __syncthreads();
    float mu  = rs[0]  * (1.0f / DMODEL);
    float var = rss[0] * (1.0f / DMODEL) - mu * mu;
    float inv = rsqrtf(var + LN_EPS);
    const float4 g  = ((const float4*)gamma)[tid];
    const float4 be = ((const float4*)beta)[tid];
    _Float16* o = xn + (size_t)row * DMODEL + tid * 4;
    o[0] = (_Float16)((v.x - mu) * inv * g.x + be.x);
    o[1] = (_Float16)((v.y - mu) * inv * g.y + be.y);
    o[2] = (_Float16)((v.z - mu) * inv * g.z + be.z);
    o[3] = (_Float16)((v.w - mu) * inv * g.w + be.w);
}

// ---------------- tiled WMMA GEMM: C[M=8192, N=1024] = A f16 * Wt^T f16 ----------------
// Wt is pre-transposed [n][k], so both A and B tiles stage with async b128 (no transpose).
// Double-buffered LDS, wave tile 32x64 (8 WMMAs per K-step, 1 barrier per K-step).
// mode 0: f16 scatter [B,H,T,DH] (Q,K). mode 1: f32 row-major + bias. mode 2: f16 [B,H,DH,T] (V).
#define BM 128
#define BN 128
#define BK 32
__global__ __launch_bounds__(256) void gemm_wmma_kernel(const _Float16* __restrict__ A,
                                                        const _Float16* __restrict__ Wt,
                                                        _Float16* __restrict__ outH,
                                                        float* __restrict__ outF,
                                                        const float* __restrict__ bias,
                                                        int mode) {
    __shared__ _Float16 As[2][BM][BK];   // 16 KB
    __shared__ _Float16 Bt[2][BN][BK];   // 16 KB ([n][k])
    int tid = threadIdx.x, lane = tid & 31, wave = tid >> 5;
    int hs = lane >> 4, ln = lane & 15;
    int m0 = blockIdx.x * BM, n0 = blockIdx.y * BN;
    int wm = (wave & 3) * 32;   // wave row tile (4 row-waves)
    int wn = (wave >> 2) * 64;  // wave col tile (2 col-waves)

    // staging indices: 128 rows x 2 segs of 16 halfs (used for both A and Bt)
    const int sar = tid >> 1, sac = (tid & 1) * 16;

    v8f acc[2][4] = {};

    // prologue: stage k0 = 0 into buffer 0
    {
        const _Float16* ap = A  + (size_t)(m0 + sar) * DMODEL + sac;
        const _Float16* wp = Wt + (size_t)(n0 + sar) * DMODEL + sac;
        ASYNC_CP16(ap,     &As[0][sar][sac]);
        ASYNC_CP16(ap + 8, &As[0][sar][sac + 8]);
        ASYNC_CP16(wp,     &Bt[0][sar][sac]);
        ASYNC_CP16(wp + 8, &Bt[0][sar][sac + 8]);
    }
    ASYNC_WAIT();
    __syncthreads();

    int p = 0;
    for (int k0 = 0; k0 < DMODEL; k0 += BK) {
        int kn = k0 + BK;
        if (kn < DMODEL) {   // stage next tile into the other buffer while computing
            const _Float16* ap = A  + (size_t)(m0 + sar) * DMODEL + kn + sac;
            const _Float16* wp = Wt + (size_t)(n0 + sar) * DMODEL + kn + sac;
            ASYNC_CP16(ap,     &As[p ^ 1][sar][sac]);
            ASYNC_CP16(ap + 8, &As[p ^ 1][sar][sac + 8]);
            ASYNC_CP16(wp,     &Bt[p ^ 1][sar][sac]);
            ASYNC_CP16(wp + 8, &Bt[p ^ 1][sar][sac + 8]);
            __builtin_prefetch(ap + BK, 0, 3);
            __builtin_prefetch(wp + BK, 0, 3);
        }

        v16h af[2], bf[4];
        for (int i = 0; i < 2; i++) {            // A fragments (16x32 f16 layout)
            int row = wm + i * 16 + ln;
            const v8h c0 = *(const v8h*)&As[p][row][hs * 8];
            const v8h c1 = *(const v8h*)&As[p][row][16 + hs * 8];
            for (int j = 0; j < 8; j++) { af[i][j] = c0[j]; af[i][8 + j] = c1[j]; }
        }
        for (int i = 0; i < 4; i++) {            // B fragments (32x16 f16 layout)
            int n = wn + i * 16 + ln;
            const v8h c0 = *(const v8h*)&Bt[p][n][hs * 8];
            const v8h c1 = *(const v8h*)&Bt[p][n][16 + hs * 8];
            for (int j = 0; j < 8; j++) { bf[i][j] = c0[j]; bf[i][8 + j] = c1[j]; }
        }
        for (int i = 0; i < 2; i++)
            for (int j = 0; j < 4; j++)
                acc[i][j] = __builtin_amdgcn_wmma_f32_16x16x32_f16(
                    false, af[i], false, bf[j], (short)0, acc[i][j], false, false);

        ASYNC_WAIT();
        __syncthreads();
        p ^= 1;
    }

    for (int i = 0; i < 2; i++)
        for (int j = 0; j < 4; j++)
            for (int r = 0; r < 8; r++) {
                int m = m0 + wm + i * 16 + r + 8 * hs;
                int n = n0 + wn + j * 16 + ln;
                float val = acc[i][j][r];
                int b = m >> 11, t = m & (TSEQ - 1);
                int h = n >> 6, dh = n & (DHEAD - 1);
                if (mode == 0) {        // [B,H,T,DH]
                    outH[(((size_t)b * NHEAD + h) * TSEQ + t) * DHEAD + dh] = (_Float16)val;
                } else if (mode == 2) { // [B,H,DH,T]  (V transposed for flash B-frags)
                    outH[(((size_t)b * NHEAD + h) * DHEAD + dh) * TSEQ + t] = (_Float16)val;
                } else {
                    outF[(size_t)m * DMODEL + n] = val + bias[n];
                }
            }
}

// ---------------- flash attention: per wave one 16-row q tile ----------------
// grid = (T/128, B*H), block = 256. K/V tiles double-buffered in LDS, 1 barrier/tile.
// V arrives pre-transposed [B,H,DH,T] so all staging is straight async b128 copies.
__global__ __launch_bounds__(256) void flash_kernel(const _Float16* __restrict__ Q,
                                                    const _Float16* __restrict__ K,
                                                    const _Float16* __restrict__ VT,
                                                    const float* __restrict__ mask,
                                                    _Float16* __restrict__ AO) {
    __shared__ _Float16 Kt[2][32][DHEAD];   // 8 KB: K tile row-major [kk][dh]
    __shared__ _Float16 Vt[2][DHEAD][32];   // 8 KB: V tile [dh][kk] (already transposed)
    __shared__ _Float16 Pl[8][16][32];      // 8 KB: per-wave P scratch (no WG barrier needed)
    int tid = threadIdx.x, lane = tid & 31, wave = tid >> 5;
    int hs = lane >> 4, ln = lane & 15;
    int bh = blockIdx.y;                    // b*16 + h
    int b = bh >> 4, h = bh & 15;
    int qt = blockIdx.x * 128 + wave * 16;

    const _Float16* Qb = Q  + ((size_t)bh * TSEQ + qt) * DHEAD;
    const _Float16* Kb = K  + (size_t)bh * TSEQ * DHEAD;
    const _Float16* Vb = VT + (size_t)bh * DHEAD * TSEQ;   // [dh][t]

    // staging indices: K: 32 rows x 8 segs of 8; V: 64 rows x 4 segs of 8
    const int skk = tid >> 3, snn = (tid & 7) * 8;
    const int svr = tid >> 2, svc = (tid & 3) * 8;

    // Q A-fragments for the two K=32 halves of DH=64 (kept in registers)
    v16h qa[2];
    for (int w = 0; w < 2; w++) {
        const _Float16* rowp = Qb + ln * DHEAD + w * 32 + hs * 8;
        const v8h c0 = *(const v8h*)rowp;
        const v8h c1 = *(const v8h*)(rowp + 16);
        for (int j = 0; j < 8; j++) { qa[w][j] = c0[j]; qa[w][8 + j] = c1[j]; }
    }

    float mrun[8], lrun[8];
    for (int r = 0; r < 8; r++) { mrun[r] = -1e30f; lrun[r] = 0.f; }
    v8f o[4] = {};

    // prologue: stage tile 0 into buffer 0
    ASYNC_CP16(Kb + (size_t)skk * DHEAD + snn, &Kt[0][skk][snn]);
    ASYNC_CP16(Vb + (size_t)svr * TSEQ + svc,  &Vt[0][svr][svc]);
    ASYNC_WAIT();
    __syncthreads();

    int p = 0;
    for (int kt = 0; kt < TSEQ / 32; kt++) {
        int k0 = kt * 32;
        if (kt + 1 < TSEQ / 32) {   // stage next 32-key tile
            int kn = k0 + 32;
            ASYNC_CP16(Kb + (size_t)(kn + skk) * DHEAD + snn, &Kt[p ^ 1][skk][snn]);
            ASYNC_CP16(Vb + (size_t)svr * TSEQ + kn + svc,    &Vt[p ^ 1][svr][svc]);
        }

        // S = Q * K^T for two 16-key subtiles (B frags: contiguous K rows from LDS)
        v8f s[2];
        for (int sub = 0; sub < 2; sub++) {
            const _Float16* kr = &Kt[p][sub * 16 + ln][0];
            v16h bf0, bf1;
            {
                const v8h c0 = *(const v8h*)(kr + hs * 8);
                const v8h c1 = *(const v8h*)(kr + 16 + hs * 8);
                for (int j = 0; j < 8; j++) { bf0[j] = c0[j]; bf0[8 + j] = c1[j]; }
            }
            {
                const v8h c0 = *(const v8h*)(kr + 32 + hs * 8);
                const v8h c1 = *(const v8h*)(kr + 32 + 16 + hs * 8);
                for (int j = 0; j < 8; j++) { bf1[j] = c0[j]; bf1[8 + j] = c1[j]; }
            }
            v8f t = {};
            t = __builtin_amdgcn_wmma_f32_16x16x32_f16(false, qa[0], false, bf0,
                                                       (short)0, t, false, false);
            t = __builtin_amdgcn_wmma_f32_16x16x32_f16(false, qa[1], false, bf1,
                                                       (short)0, t, false, false);
            s[sub] = t;
        }

        // scale, multiplicative mask, online softmax (row stats across 16 lanes)
        float alpha[8], e0[8], e1[8];
        for (int r = 0; r < 8; r++) {
            int row = qt + r + 8 * hs;
            float a0 = s[0][r] * SCALE * mask[(size_t)row * TSEQ + k0 + ln];
            float a1 = s[1][r] * SCALE * mask[(size_t)row * TSEQ + k0 + 16 + ln];
            float mx = fmaxf(a0, a1);
            for (int off = 1; off < 16; off <<= 1) mx = fmaxf(mx, __shfl_xor(mx, off, 16));
            float mnew = fmaxf(mrun[r], mx);
            alpha[r] = __expf(mrun[r] - mnew);
            e0[r] = __expf(a0 - mnew);
            e1[r] = __expf(a1 - mnew);
            float rsum = e0[r] + e1[r];
            for (int off = 1; off < 16; off <<= 1) rsum += __shfl_xor(rsum, off, 16);
            lrun[r] = lrun[r] * alpha[r] + rsum;
            mrun[r] = mnew;
        }
        for (int j = 0; j < 4; j++)
            for (int r = 0; r < 8; r++) o[j][r] *= alpha[r];

        // P (C layout) -> per-wave LDS scratch -> A fragment layout.
        // Same-wave DS ops are in-order; no workgroup barrier required here.
        for (int r = 0; r < 8; r++) {
            Pl[wave][r + 8 * hs][ln]      = (_Float16)e0[r];
            Pl[wave][r + 8 * hs][16 + ln] = (_Float16)e1[r];
        }
        v16h pa;
        {
            const v8h c0 = *(const v8h*)&Pl[wave][ln][hs * 8];
            const v8h c1 = *(const v8h*)&Pl[wave][ln][16 + hs * 8];
            for (int j = 0; j < 8; j++) { pa[j] = c0[j]; pa[8 + j] = c1[j]; }
        }
        // O += P * V  (4 column tiles of DH)
        for (int j = 0; j < 4; j++) {
            int n = j * 16 + ln;
            v16h bv;
            const v8h c0 = *(const v8h*)&Vt[p][n][hs * 8];
            const v8h c1 = *(const v8h*)&Vt[p][n][16 + hs * 8];
            for (int q2 = 0; q2 < 8; q2++) { bv[q2] = c0[q2]; bv[8 + q2] = c1[q2]; }
            o[j] = __builtin_amdgcn_wmma_f32_16x16x32_f16(false, pa, false, bv,
                                                          (short)0, o[j], false, false);
        }

        ASYNC_WAIT();
        __syncthreads();
        p ^= 1;
    }

    // finalize: O/l -> AO[B,T, h*64+dh] f16
    for (int r = 0; r < 8; r++) {
        float inv = 1.0f / lrun[r];
        int t = qt + r + 8 * hs;
        size_t base = ((size_t)b * TSEQ + t) * DMODEL + h * DHEAD;
        for (int j = 0; j < 4; j++)
            AO[base + j * 16 + ln] = (_Float16)(o[j][r] * inv);
    }
}

extern "C" void kernel_launch(void* const* d_in, const int* in_sizes, int n_in,
                              void* d_out, int out_size, void* d_ws, size_t ws_size,
                              hipStream_t stream) {
    const float* x     = (const float*)d_in[0];
    const float* mask  = (const float*)d_in[1];
    const float* gamma = (const float*)d_in[2];
    const float* beta  = (const float*)d_in[3];
    const float* Wq    = (const float*)d_in[4];
    const float* Wk    = (const float*)d_in[5];
    const float* Wv    = (const float*)d_in[6];
    const float* Wo    = (const float*)d_in[7];
    const float* bo    = (const float*)d_in[8];
    float* out = (float*)d_out;

    char* ws = (char*)d_ws;
    _Float16* Wq_t = (_Float16*)(ws + OFF_WQ);
    _Float16* Wk_t = (_Float16*)(ws + OFF_WK);
    _Float16* Wv_t = (_Float16*)(ws + OFF_WV);
    _Float16* Wo_t = (_Float16*)(ws + OFF_WO);
    _Float16* xn_h = (_Float16*)(ws + OFF_XN);
    _Float16* Q_h  = (_Float16*)(ws + OFF_Q);
    _Float16* K_h  = (_Float16*)(ws + OFF_K);
    _Float16* VT_h = (_Float16*)(ws + OFF_V);
    _Float16* AO_h = (_Float16*)(ws + OFF_AO);

    dim3 tg(DMODEL / 64, DMODEL / 64);
    transpose_cast_kernel<<<tg, 256, 0, stream>>>(Wq, Wq_t);
    transpose_cast_kernel<<<tg, 256, 0, stream>>>(Wk, Wk_t);
    transpose_cast_kernel<<<tg, 256, 0, stream>>>(Wv, Wv_t);
    transpose_cast_kernel<<<tg, 256, 0, stream>>>(Wo, Wo_t);

    ln_kernel<<<MROWS, 256, 0, stream>>>(x, gamma, beta, xn_h);

    dim3 gg(MROWS / BM, DMODEL / BN);
    gemm_wmma_kernel<<<gg, 256, 0, stream>>>(xn_h, Wq_t, Q_h, nullptr, nullptr, 0);
    gemm_wmma_kernel<<<gg, 256, 0, stream>>>(xn_h, Wk_t, K_h, nullptr, nullptr, 0);
    gemm_wmma_kernel<<<gg, 256, 0, stream>>>(xn_h, Wv_t, VT_h, nullptr, nullptr, 2);

    dim3 fg(TSEQ / 128, BATCH * NHEAD);
    flash_kernel<<<fg, 256, 0, stream>>>(Q_h, K_h, VT_h, mask, AO_h);

    gemm_wmma_kernel<<<gg, 256, 0, stream>>>(AO_h, Wo_t, nullptr, out, bo, 1);
}
